// MARNN_72773925863758
// MI455X (gfx1250) — matrix operand order repaired and checked by reading, not compiled, not run on hardware
//
#include <hip/hip_runtime.h>

// ---------------------------------------------------------------------------
// MARNN on gfx1250: all GEMMs via v_wmma_f32_16x16x32_bf16 (wave32).
// Shapes: T=256, B=128, X=256, H=512, R=128, M=64.  K dims: X+H=768, X+H+R=896.
// Large GEMMs: 2x2 register blocking (4 WMMAs per 8 b128 loads), k-loop fully
// unrolled so the backend folds k offsets into load immediates and pipelines.
// ---------------------------------------------------------------------------

typedef __bf16 bf16_t;
typedef bf16_t v16bf __attribute__((ext_vector_type(16)));
typedef float  v8f   __attribute__((ext_vector_type(8)));

#define T_N 256
#define B_N 128
#define X_N 256
#define H_N 512
#define R_N 128
#define M_N 64

__device__ __forceinline__ unsigned short f2bf(float f) {
    unsigned int u = __float_as_uint(f);
    unsigned int r = (u + 0x7FFFu + ((u >> 16) & 1u)) >> 16;  // RNE
    return (unsigned short)r;
}
__device__ __forceinline__ float sigm(float x) { return 1.0f / (1.0f + __expf(-x)); }

// ---------------------------------------------------------------------------
// Fragment layouts (see CDNA5 ISA):
// A (16x32 bf16): lane row = (lane&15); half = lane>>4.
//   slots 0..7  = K [k0 + 8*half, +8), slots 8..15 = K [k0+16+8*half, +8).
// B pre-packed per (kt,nt) tile: 32 lanes x 16 bf16 contiguous;
//   lane slot s = B[K = kt*32 + 8*(lane>>4) + s + 8*(s>>3)][N = nt*16 + (lane&15)].
// C/D (16x16 f32): elem r -> row r + 8*(lane>>4), col lane&15.
// ---------------------------------------------------------------------------

// Single-tile GEMM (small GEMMs), fully unrolled.
template <int KT>
__device__ __forceinline__ v8f gemm_tile_bf16(
    const unsigned short* __restrict__ A, int lda,
    const unsigned short* __restrict__ Bp, int nTilesN,
    int mTile, int nTile, int lane)
{
    v8f acc = {0.f, 0.f, 0.f, 0.f, 0.f, 0.f, 0.f, 0.f};
    const int half = lane >> 4;
    const int row  = mTile * 16 + (lane & 15);
    const unsigned short* arow = A + (size_t)row * lda + 8 * half;
    const unsigned short* bp =
        Bp + (size_t)nTile * 512 + (size_t)lane * 16;
    const size_t bstride = (size_t)nTilesN * 512;
#pragma unroll
    for (int kt = 0; kt < KT; ++kt) {
        union { uint4 q[2]; v16bf v; } af, bfv;
        af.q[0]  = *(const uint4*)(arow + kt * 32);
        af.q[1]  = *(const uint4*)(arow + kt * 32 + 16);
        bfv.q[0] = *(const uint4*)(bp + kt * bstride);
        bfv.q[1] = *(const uint4*)(bp + kt * bstride + 8);
        acc = __builtin_amdgcn_wmma_f32_16x16x32_bf16(
            false, af.v, false, bfv.v, (short)0, acc, false, false);
    }
    return acc;
}

// 2x2-blocked GEMM: wave computes M-tiles {2mB,2mB+1} x N-tiles {2nB,2nB+1}.
// A/B fragments each reused twice -> halves L2 fetch per WMMA, 4-way ILP.
template <int KT>
__device__ __forceinline__ void gemm_tile2x2_bf16(
    const unsigned short* __restrict__ A, int lda,
    const unsigned short* __restrict__ Bp, int nTilesN,
    int mBlk, int nBlk, int lane, v8f acc[4])
{
    acc[0] = acc[1] = acc[2] = acc[3] = (v8f){0.f,0.f,0.f,0.f,0.f,0.f,0.f,0.f};
    const int half = lane >> 4;
    const int row0 = (2 * mBlk) * 16 + (lane & 15);
    const unsigned short* arow0 = A + (size_t)row0 * lda + 8 * half;
    const unsigned short* arow1 = arow0 + (size_t)16 * lda;
    const unsigned short* bp0 =
        Bp + (size_t)(2 * nBlk) * 512 + (size_t)lane * 16;
    const size_t bstride = (size_t)nTilesN * 512;
#pragma unroll
    for (int kt = 0; kt < KT; ++kt) {
        union { uint4 q[2]; v16bf v; } a0, a1, b0, b1;
        a0.q[0] = *(const uint4*)(arow0 + kt * 32);
        a0.q[1] = *(const uint4*)(arow0 + kt * 32 + 16);
        a1.q[0] = *(const uint4*)(arow1 + kt * 32);
        a1.q[1] = *(const uint4*)(arow1 + kt * 32 + 16);
        const unsigned short* bp = bp0 + kt * bstride;
        b0.q[0] = *(const uint4*)(bp);
        b0.q[1] = *(const uint4*)(bp + 8);
        b1.q[0] = *(const uint4*)(bp + 512);
        b1.q[1] = *(const uint4*)(bp + 520);
        acc[0] = __builtin_amdgcn_wmma_f32_16x16x32_bf16(
            false, a0.v, false, b0.v, (short)0, acc[0], false, false);
        acc[1] = __builtin_amdgcn_wmma_f32_16x16x32_bf16(
            false, a0.v, false, b1.v, (short)0, acc[1], false, false);
        acc[2] = __builtin_amdgcn_wmma_f32_16x16x32_bf16(
            false, a1.v, false, b0.v, (short)0, acc[2], false, false);
        acc[3] = __builtin_amdgcn_wmma_f32_16x16x32_bf16(
            false, a1.v, false, b1.v, (short)0, acc[3], false, false);
    }
}

// ---------------------------------------------------------------------------
// Prologue kernels
// ---------------------------------------------------------------------------
__global__ __launch_bounds__(256) void k_init_c(const float* __restrict__ c_bias,
                                                float* __restrict__ c) {
    int idx = blockIdx.x * 256 + threadIdx.x;           // B*H = 65536
    c[idx] = tanhf(c_bias[idx & (H_N - 1)]);
}

__global__ __launch_bounds__(256) void k_init_hmem(const float* __restrict__ hb,
                                                   float* __restrict__ hmem) {
    int idx = blockIdx.x * 256 + threadIdx.x;           // B*M*R = 1048576
    hmem[idx] = hb[idx & (M_N * R_N - 1)];
}

// Pack row-major f32 KxN weight -> bf16 WMMA-B tiles (32x16 per tile).
__global__ __launch_bounds__(256) void k_pack(const float* __restrict__ W,
                                              unsigned short* __restrict__ Wp,
                                              int N, int total) {
    int idx = blockIdx.x * 256 + threadIdx.x;
    if (idx >= total) return;
    int nTiles = N >> 4;
    int tile = idx >> 9;
    int within = idx & 511;
    int lane = within >> 4;
    int slot = within & 15;
    int kt = tile / nTiles;
    int nt = tile - kt * nTiles;
    int k = kt * 32 + ((lane >> 4) << 3) + slot + ((slot >> 3) << 3);
    int n = nt * 16 + (lane & 15);
    Wp[idx] = f2bf(W[(size_t)k * N + n]);
}

// ---------------------------------------------------------------------------
// Per-step kernels
// ---------------------------------------------------------------------------

// K0: stage x_t and c into bf16 activation buffers. grid = B*768 threads.
__global__ __launch_bounds__(256) void k_prep_x(
    const float* __restrict__ x, const float* __restrict__ c,
    unsigned short* __restrict__ concatbf, unsigned short* __restrict__ gatedbf,
    unsigned short* __restrict__ xcnew, int t)
{
    int idx = blockIdx.x * 256 + threadIdx.x;
    int b = idx / 768, j = idx - b * 768;
    if (j < X_N) {
        unsigned short bf = f2bf(x[((size_t)t * B_N + b) * X_N + j]);
        concatbf[b * 896 + j] = bf;   // concat = [x, c, h_entry]
        gatedbf[b * 896 + j]  = bf;   // gate for x-cols is 1.0
        xcnew[b * 768 + j]    = bf;   // [x, new_c] for trans GEMM
    } else {
        concatbf[b * 896 + j] = f2bf(c[b * H_N + (j - X_N)]);
    }
}

// K1: head = [x,c] @ fc_w + fc_b.  M=128, K=768, N=64 -> 8x4 = 32 waves.
__global__ __launch_bounds__(256) void k_gemm_head(
    const unsigned short* __restrict__ A, const unsigned short* __restrict__ Bp,
    const float* __restrict__ bias, float* __restrict__ out)
{
    int tid = threadIdx.x, lane = tid & 31;
    int wid = blockIdx.x * 8 + (tid >> 5);
    int mT = wid & 7, nT = wid >> 3;
    v8f acc = gemm_tile_bf16<24>(A, 896, Bp, 4, mT, nT, lane);
    int half = lane >> 4, col = nT * 16 + (lane & 15);
    float bv = bias[col];
#pragma unroll
    for (int r = 0; r < 8; ++r) {
        int row = mT * 16 + r + 8 * half;
        out[row * 64 + col] = acc[r] + bv;
    }
}

// K1b: softmax over M=64, h_entry = h_index @ hmem[b]. One block per batch row.
__global__ __launch_bounds__(128) void k_softmax_entry(
    const float* __restrict__ head, const float* __restrict__ hmem,
    float* __restrict__ hidx, float* __restrict__ hentry,
    unsigned short* __restrict__ concatbf)
{
    int b = blockIdx.x, tid = threadIdx.x;
    __shared__ float sp[M_N];
    __shared__ float sred;
    if (tid < M_N) sp[tid] = head[b * M_N + tid];
    __syncthreads();
    if (tid == 0) {
        float mx = sp[0];
        for (int m = 1; m < M_N; ++m) mx = fmaxf(mx, sp[m]);
        sred = mx;
    }
    __syncthreads();
    float mx = sred;
    if (tid < M_N) sp[tid] = __expf(sp[tid] - mx);
    __syncthreads();
    if (tid == 0) {
        float s = 0.f;
        for (int m = 0; m < M_N; ++m) s += sp[m];
        sred = s;
    }
    __syncthreads();
    float inv = 1.0f / sred;
    if (tid < M_N) {
        float p = sp[tid] * inv;
        sp[tid] = p;
        hidx[b * M_N + tid] = p;
    }
    __syncthreads();
    int r = tid;  // 0..127
    float acc = 0.f;
    const float* hb = hmem + (size_t)b * (M_N * R_N) + r;
#pragma unroll 4
    for (int m = 0; m < M_N; ++m) acc += sp[m] * hb[m * R_N];
    hentry[b * R_N + r] = acc;
    concatbf[b * 896 + 768 + r] = f2bf(acc);
}

// K2: gate = sigmoid(concat @ W_full1 + b1); gated cols 256.. = concat*gate.
// M=128, K=896, N=640.  2x2 blocked: 4x20 = 80 waves.
__global__ __launch_bounds__(256) void k_gemm_gate(
    const unsigned short* __restrict__ A, const unsigned short* __restrict__ Bp,
    const float* __restrict__ bias1, const float* __restrict__ c,
    const float* __restrict__ hentry, unsigned short* __restrict__ gatedbf)
{
    int tid = threadIdx.x, lane = tid & 31;
    int wid = blockIdx.x * 8 + (tid >> 5);
    int mB = wid & 3, nB = wid >> 2;        // nB in [0,20)
    v8f acc[4];
    gemm_tile2x2_bf16<28>(A, 896, Bp, 40, mB, nB, lane, acc);
    int half = lane >> 4, cl = lane & 15;
#pragma unroll
    for (int i = 0; i < 2; ++i) {
#pragma unroll
        for (int j = 0; j < 2; ++j) {
            int mT = 2 * mB + i, nT = 2 * nB + j;
            int n = nT * 16 + cl;
            float bv = bias1[n];
            v8f a = acc[i * 2 + j];
#pragma unroll
            for (int r = 0; r < 8; ++r) {
                int row = mT * 16 + r + 8 * half;
                float src = (n < H_N) ? c[row * H_N + n]
                                      : hentry[row * R_N + (n - H_N)];
                float g = sigm(a[r] + bv);
                gatedbf[row * 896 + X_N + n] = f2bf(src * g);
            }
        }
    }
}

// K3: z = gated @ W_full + bias.  M=128, K=896, N=2176.
// 2x2 blocked: 4x68 = 272 waves, 112 WMMAs each.
__global__ __launch_bounds__(256) void k_gemm_z(
    const unsigned short* __restrict__ A, const unsigned short* __restrict__ Bp,
    const float* __restrict__ bias, float* __restrict__ z)
{
    int tid = threadIdx.x, lane = tid & 31;
    int wid = blockIdx.x * 8 + (tid >> 5);
    int mB = wid & 3, nB = wid >> 2;        // nB in [0,68)
    v8f acc[4];
    gemm_tile2x2_bf16<28>(A, 896, Bp, 136, mB, nB, lane, acc);
    int half = lane >> 4, cl = lane & 15;
#pragma unroll
    for (int i = 0; i < 2; ++i) {
#pragma unroll
        for (int j = 0; j < 2; ++j) {
            int mT = 2 * mB + i, nT = 2 * nB + j;
            int col = nT * 16 + cl;
            float bv = bias[col];
            v8f a = acc[i * 2 + j];
#pragma unroll
            for (int r = 0; r < 8; ++r) {
                int row = mT * 16 + r + 8 * half;
                z[(size_t)row * 2176 + col] = a[r] + bv;
            }
        }
    }
}

// K4: LSTM cell + outputs. grid = B*(H+R) = 81920 threads.
__global__ __launch_bounds__(256) void k_cell(
    const float* __restrict__ z, float* __restrict__ c,
    const float* __restrict__ hentry, unsigned short* __restrict__ xcnew,
    float* __restrict__ out, int t)
{
    int idx = blockIdx.x * 256 + threadIdx.x;
    int b = idx / 640, j = idx - b * 640;
    float* orow = out + ((size_t)t * B_N + b) * 640;
    const float* zr = z + (size_t)b * 2176;
    if (j < H_N) {
        float zi = zr[j], zj = zr[H_N + j], zf = zr[2 * H_N + j], zo = zr[3 * H_N + j];
        float cold = c[b * H_N + j];
        float nc = tanhf(cold * sigm(zf + 1.0f) + sigm(zi) * tanhf(zj));
        float nh = nc * sigm(zo);
        c[b * H_N + j] = nc;
        xcnew[b * 768 + X_N + j] = f2bf(nc);
        orow[j] = nh;
    } else {
        int r = j - H_N;
        float om = zr[4 * H_N + r];
        orow[j] = hentry[b * R_N + r] * sigm(om);
    }
}

// K5: h_write = [x, new_c] @ trans_w + trans_b.  8x8 = 64 waves.
__global__ __launch_bounds__(256) void k_gemm_trans(
    const unsigned short* __restrict__ A, const unsigned short* __restrict__ Bp,
    const float* __restrict__ bias, float* __restrict__ hw)
{
    int tid = threadIdx.x, lane = tid & 31;
    int wid = blockIdx.x * 8 + (tid >> 5);
    int mT = wid & 7, nT = wid >> 3;
    v8f acc = gemm_tile_bf16<24>(A, 768, Bp, 8, mT, nT, lane);
    int half = lane >> 4, col = nT * 16 + (lane & 15);
    float bv = bias[col];
#pragma unroll
    for (int r = 0; r < 8; ++r) {
        int row = mT * 16 + r + 8 * half;
        hw[row * R_N + col] = acc[r] + bv;
    }
}

// K6: hmem blend. grid = B*M*R = 1048576 threads.
__global__ __launch_bounds__(256) void k_hmem(
    float* __restrict__ hmem, const float* __restrict__ hwrite,
    const float* __restrict__ hidx, int t)
{
    int idx = blockIdx.x * 256 + threadIdx.x;
    int b = idx >> 13;
    int rem = idx & 8191;
    int m = rem >> 7;
    int r = rem & 127;
    float w = (t < M_N) ? ((m == t) ? 1.0f : 0.0f) : hidx[b * M_N + m];
    float hv = hmem[idx];
    hmem[idx] = hwrite[b * R_N + r] * w + hv * (1.0f - w);
}

// ---------------------------------------------------------------------------
extern "C" void kernel_launch(void* const* d_in, const int* in_sizes, int n_in,
                              void* d_out, int out_size, void* d_ws, size_t ws_size,
                              hipStream_t stream)
{
    (void)in_sizes; (void)n_in; (void)out_size; (void)ws_size;
    const float* x         = (const float*)d_in[0];
    const float* W_full    = (const float*)d_in[1];
    const float* bias      = (const float*)d_in[2];
    const float* W_full1   = (const float*)d_in[3];
    const float* bias1     = (const float*)d_in[4];
    const float* trans_w   = (const float*)d_in[5];
    const float* trans_b   = (const float*)d_in[6];
    const float* c_bias    = (const float*)d_in[7];
    const float* hmem_bias = (const float*)d_in[8];
    const float* fc_w      = (const float*)d_in[9];
    const float* fc_b      = (const float*)d_in[10];
    float* out = (float*)d_out;

    char* base = (char*)d_ws;
    size_t off = 0;
    auto take = [&](size_t bytes) -> char* {
        char* p = base + off;
        off = (off + bytes + 255) & ~(size_t)255;
        return p;
    };
    float* c_f32            = (float*)take((size_t)B_N * H_N * 4);
    float* hmem             = (float*)take((size_t)B_N * M_N * R_N * 4);
    unsigned short* Wpack   = (unsigned short*)take((size_t)896 * 2176 * 2);
    unsigned short* W1pack  = (unsigned short*)take((size_t)896 * 640 * 2);
    unsigned short* fcpack  = (unsigned short*)take((size_t)768 * 64 * 2);
    unsigned short* tpack   = (unsigned short*)take((size_t)768 * 128 * 2);
    unsigned short* concatbf= (unsigned short*)take((size_t)B_N * 896 * 2);
    unsigned short* gatedbf = (unsigned short*)take((size_t)B_N * 896 * 2);
    unsigned short* xcnew   = (unsigned short*)take((size_t)B_N * 768 * 2);
    float* headb            = (float*)take((size_t)B_N * M_N * 4);
    float* hidx             = (float*)take((size_t)B_N * M_N * 4);
    float* hentry           = (float*)take((size_t)B_N * R_N * 4);
    float* zbuf             = (float*)take((size_t)B_N * 2176 * 4);
    float* hwrite           = (float*)take((size_t)B_N * R_N * 4);

    // Prologue: state init + weight packing (deterministic; runs every call).
    k_init_c<<<256, 256, 0, stream>>>(c_bias, c_f32);
    k_init_hmem<<<4096, 256, 0, stream>>>(hmem_bias, hmem);
    k_pack<<<7616, 256, 0, stream>>>(W_full, Wpack, 2176, 896 * 2176);
    k_pack<<<2240, 256, 0, stream>>>(W_full1, W1pack, 640, 896 * 640);
    k_pack<<<192, 256, 0, stream>>>(fc_w, fcpack, 64, 768 * 64);
    k_pack<<<384, 256, 0, stream>>>(trans_w, tpack, 128, 768 * 128);

    for (int t = 0; t < T_N; ++t) {
        k_prep_x<<<384, 256, 0, stream>>>(x, c_f32, concatbf, gatedbf, xcnew, t);
        k_gemm_head<<<4, 256, 0, stream>>>(concatbf, fcpack, fc_b, headb);
        k_softmax_entry<<<128, 128, 0, stream>>>(headb, hmem, hidx, hentry, concatbf);
        k_gemm_gate<<<10, 256, 0, stream>>>(concatbf, W1pack, bias1, c_f32, hentry, gatedbf);
        k_gemm_z<<<34, 256, 0, stream>>>(gatedbf, Wpack, bias, zbuf);
        k_cell<<<320, 256, 0, stream>>>(zbuf, c_f32, hentry, xcnew, out, t);
        k_gemm_trans<<<8, 256, 0, stream>>>(xcnew, tpack, trans_b, hwrite);
        k_hmem<<<4096, 256, 0, stream>>>(hmem, hwrite, hidx, t);
    }
}
